// SimpleAttention_73529840107636
// MI455X (gfx1250) — compile-verified, matrix operand order checked by reading, not run on hardware
//
#include <hip/hip_runtime.h>

#define DM  1024
#define NH  16
#define HD  64
#define BSZ 2
#define SEQ 2048
#define MTOT (BSZ * SEQ)   // 4096 rows
#define KTILES (SEQ / 64)  // 32 key tiles

typedef __bf16 v16bf __attribute__((ext_vector_type(16)));
typedef float  v8f   __attribute__((ext_vector_type(8)));
typedef float  f32x4 __attribute__((ext_vector_type(4)));
typedef unsigned int u32x4 __attribute__((ext_vector_type(4)));
typedef unsigned int v4u  __attribute__((ext_vector_type(4)));
typedef int          v4i  __attribute__((ext_vector_type(4)));
typedef int          v8i  __attribute__((ext_vector_type(8)));

// ---------------- TDM (Tensor Data Mover) support, toolchain-guarded --------
#if defined(__has_builtin)
#if __has_builtin(__builtin_amdgcn_tensor_load_to_lds)
#define HAVE_TDM 1
#if __has_include(<hip/amd_detail/amd_gfx1250_TDM.h>)
#define TDM_CALL(g0, g1, g2, g3) \
    __builtin_amdgcn_tensor_load_to_lds((g0), (g1), (g2), (g3), (v8i){0,0,0,0,0,0,0,0}, 0)
#else
#define TDM_CALL(g0, g1, g2, g3) \
    __builtin_amdgcn_tensor_load_to_lds((g0), (g1), (g2), (g3), 0)
#endif
#endif
#endif
#ifndef HAVE_TDM
#define HAVE_TDM 0
#endif

#if HAVE_TDM
static __device__ __forceinline__ void tdm_wait_tensor0() {
#if __has_builtin(__builtin_amdgcn_s_wait_tensorcnt)
    __builtin_amdgcn_s_wait_tensorcnt(0);
#else
    asm volatile("s_wait_tensorcnt 0x0" ::: "memory");
#endif
}

// 2D tile DMA: tile_w x tile_h elements (2B), row stride in elements.
// Pad fields add +16B every 128B so a 64-elem row lands on a 72-halfword pitch.
static __device__ __forceinline__ void tdm_load_2d_bf16(
    const void* gsrc, unsigned lds_byte_addr,
    unsigned tile_w, unsigned tile_h, unsigned row_stride)
{
    unsigned long long ga = (unsigned long long)(size_t)gsrc;
    v4u g0;
    g0.x = 1u;                                   // count=1, user descriptor
    g0.y = lds_byte_addr;
    g0.z = (unsigned)ga;
    g0.w = (unsigned)((ga >> 32) & 0x01FFFFFFu) | (2u << 30);   // type=2
    v8i g1;
    g1[0] = (int)((1u << 16) | (1u << 20) | (4u << 22) | (3u << 25));
    g1[1] = (int)((tile_w & 0xFFFFu) << 16);                    // tensor_dim0 lo
    g1[2] = (int)((tile_w >> 16) | ((tile_h & 0xFFFFu) << 16)); // dim0 hi | dim1 lo
    g1[3] = (int)((tile_h >> 16) | ((tile_w & 0xFFFFu) << 16)); // dim1 hi | tile_dim0
    g1[4] = (int)(tile_h & 0xFFFFu);                            // tile_dim1
    g1[5] = (int)row_stride;                                    // dim0_stride lo32
    g1[6] = 0;
    g1[7] = 0;
    v4i z = {0, 0, 0, 0};
    TDM_CALL(g0, g1, z, z);
}
#endif

// float -> bf16 round-to-nearest-even
static __device__ __forceinline__ unsigned short f2bf(float f) {
    unsigned int u = __float_as_uint(f);
    u += 0x7FFFu + ((u >> 16) & 1u);
    return (unsigned short)(u >> 16);
}

// Gather one 16-bit A/B WMMA fragment from an LDS row (two 16B loads per lane).
static __device__ __forceinline__ v16bf frag_ld(const unsigned short* rowBase, int half) {
    union { u32x4 q[2]; v16bf v; } u;
    u.q[0] = *(const u32x4*)(rowBase + half * 8);
    u.q[1] = *(const u32x4*)(rowBase + 16 + half * 8);
    return u.v;
}

static __device__ __forceinline__ v8f wmma_bf16(v16bf a, v16bf b, v8f c) {
    return __builtin_amdgcn_wmma_f32_16x16x32_bf16(false, a, false, b, (short)0, c, false, false);
}

// ---------------------------------------------------------------------------
// x (f32) -> bf16 streaming conversion, 8 elements/thread.
// ---------------------------------------------------------------------------
__global__ __launch_bounds__(256) void f32_to_bf16_kernel(
    const float* __restrict__ in, unsigned short* __restrict__ out)
{
    const int i = (blockIdx.x * 256 + threadIdx.x) * 8;
    f32x4 a = *(const f32x4*)(in + i);
    f32x4 b = *(const f32x4*)(in + i + 4);
    u32x4 p;
    p.x = (unsigned)f2bf(a.x) | ((unsigned)f2bf(a.y) << 16);
    p.y = (unsigned)f2bf(a.z) | ((unsigned)f2bf(a.w) << 16);
    p.z = (unsigned)f2bf(b.x) | ((unsigned)f2bf(b.y) << 16);
    p.w = (unsigned)f2bf(b.z) | ((unsigned)f2bf(b.w) << 16);
    *(u32x4*)(out + i) = p;
}

// ---------------------------------------------------------------------------
// Tiled GEMM: C[M,N] = A[M,K](bf16) @ W[K,N](f32) + bias. BM=128 BN=64 BK=32.
// Double-buffered LDS, single barrier per K-step; next tile's global loads
// are issued right after the barrier and overlap the WMMA work.
// OUT_MODE 0: float out[m*DM+n].  OUT_MODE 1: bf16 scatter to [B,H,S,Hd].
// ---------------------------------------------------------------------------
template <int OUT_MODE>
__global__ __launch_bounds__(256) void gemm_bias_kernel(
    const unsigned short* __restrict__ A, const float* __restrict__ W,
    const float* __restrict__ bias, void* __restrict__ Outp)
{
    __shared__ __attribute__((aligned(16))) unsigned short ldsA[2 * 128 * 40]; // [m][k]
    __shared__ __attribute__((aligned(16))) unsigned short ldsB[2 * 64 * 40];  // [n][k]

    const int t    = threadIdx.x;
    const int wave = t >> 5, lane = t & 31;
    const int lm   = lane & 15, half = lane >> 4;
    const int wm   = wave >> 1, wn = wave & 1;
    const int tileN = blockIdx.x * 64;
    const int tileM = blockIdx.y * 128;

    const int ar = t >> 1, ac = (t & 1) * 16;   // A tile 128x32: 16 elems/thread
    const int bk = t >> 3, bn = (t & 7) * 8;    // B tile 32x64:  8 elems/thread

    v8f acc[2][2];
    acc[0][0] = {}; acc[0][1] = {}; acc[1][0] = {}; acc[1][1] = {};

    // prologue: load tile 0 into registers
    u32x4 ra0, ra1; f32x4 rb0, rb1;
    {
        const unsigned short* s = A + (size_t)(tileM + ar) * DM + ac;
        ra0 = *(const u32x4*)s; ra1 = *(const u32x4*)(s + 8);
        const float* sw = W + (size_t)bk * DM + tileN + bn;
        rb0 = *(const f32x4*)sw; rb1 = *(const f32x4*)(sw + 4);
    }

    for (int it = 0; it < DM / 32; ++it) {
        unsigned short* Ab = ldsA + (it & 1) * (128 * 40);
        unsigned short* Bb = ldsB + (it & 1) * (64 * 40);
        // ---- store staged registers to LDS ----
        *(u32x4*)&Ab[ar * 40 + ac]     = ra0;
        *(u32x4*)&Ab[ar * 40 + ac + 8] = ra1;
        {
            unsigned short* dst = &Bb[bn * 40 + bk];
            dst[0]   = f2bf(rb0.x); dst[40]  = f2bf(rb0.y);
            dst[80]  = f2bf(rb0.z); dst[120] = f2bf(rb0.w);
            dst[160] = f2bf(rb1.x); dst[200] = f2bf(rb1.y);
            dst[240] = f2bf(rb1.z); dst[280] = f2bf(rb1.w);
        }
        __syncthreads();   // s_wait_dscnt 0 + barrier: also drains prev tile reads
        // ---- issue next tile's global loads (overlap with compute) ----
        if (it + 1 < DM / 32) {
            const int kt = (it + 1) * 32;
            const unsigned short* s = A + (size_t)(tileM + ar) * DM + kt + ac;
            ra0 = *(const u32x4*)s; ra1 = *(const u32x4*)(s + 8);
            const float* sw = W + (size_t)(kt + bk) * DM + tileN + bn;
            rb0 = *(const f32x4*)sw; rb1 = *(const f32x4*)(sw + 4);
        }
        // ---- compute: 4 WMMAs (32x32x32) per wave ----
        v16bf af0 = frag_ld(&Ab[(wm * 32 + lm) * 40], half);
        v16bf af1 = frag_ld(&Ab[(wm * 32 + 16 + lm) * 40], half);
        v16bf bf0 = frag_ld(&Bb[(wn * 32 + lm) * 40], half);
        v16bf bf1 = frag_ld(&Bb[(wn * 32 + 16 + lm) * 40], half);
        acc[0][0] = wmma_bf16(af0, bf0, acc[0][0]);
        acc[0][1] = wmma_bf16(af0, bf1, acc[0][1]);
        acc[1][0] = wmma_bf16(af1, bf0, acc[1][0]);
        acc[1][1] = wmma_bf16(af1, bf1, acc[1][1]);
    }

    // ---- epilogue: bias + store ----
    #pragma unroll
    for (int mi = 0; mi < 2; ++mi) {
        #pragma unroll
        for (int ni = 0; ni < 2; ++ni) {
            const int ncol = tileN + wn * 32 + ni * 16 + lm;
            const float bv = bias[ncol];
            #pragma unroll
            for (int v = 0; v < 8; ++v) {
                const int mrow = tileM + wm * 32 + mi * 16 + v + half * 8;
                const float val = acc[mi][ni][v] + bv;
                if (OUT_MODE == 0) {
                    ((float*)Outp)[(size_t)mrow * DM + ncol] = val;
                } else {
                    const int b = mrow >> 11, s = mrow & (SEQ - 1);
                    const int h = ncol >> 6,  d = ncol & (HD - 1);
                    ((unsigned short*)Outp)[(((size_t)(b * NH + h)) * SEQ + s) * HD + d] =
                        f2bf(val);
                }
            }
        }
    }
}

// ---------------------------------------------------------------------------
// Flash attention, pipelined. Block = 128 queries x 1 head; 8 waves.
// K tiles: TDM DMA into double-buffered LDS, prefetched one tile ahead.
// V tiles: global->register prefetch, transposed into LDS at use time.
// ---------------------------------------------------------------------------
__global__ __launch_bounds__(256) void attn_kernel(
    const unsigned short* __restrict__ Q, const unsigned short* __restrict__ K,
    const unsigned short* __restrict__ V, unsigned short* __restrict__ O)
{
    __shared__ __attribute__((aligned(16))) unsigned short ldsQ[128 * 72];    // [q][d]
    __shared__ __attribute__((aligned(16))) unsigned short ldsK[2 * 64 * 72]; // [key][d]
    __shared__ __attribute__((aligned(16))) unsigned short ldsV[2 * 64 * 72]; // [d][key]
    __shared__ __attribute__((aligned(16))) unsigned short ldsP[8 * 16 * 72];

    const int t    = threadIdx.x;
    const int wave = t >> 5, lane = t & 31;
    const int lm   = lane & 15, half = lane >> 4;
    const int qb   = blockIdx.x;
    const int bh   = blockIdx.y;

    const size_t head_off = (size_t)bh * SEQ * HD;
    const unsigned short* Qh = Q + head_off + (size_t)qb * 128 * HD;
    const unsigned short* Kh = K + head_off;
    const unsigned short* Vh = V + head_off;

    const int vr = t >> 2, vc = (t & 3) * 16;   // 64x64 tile: 16 elems/thread

    // ---- prologue: stage Q (TDM) + issue K tile 0 DMA + load V tile 0 regs ----
#if HAVE_TDM
    if (wave == 0) {
        tdm_load_2d_bf16(Qh, (unsigned)(size_t)(const void*)&ldsQ[0], HD, 128, HD);
        tdm_load_2d_bf16(Kh, (unsigned)(size_t)(const void*)&ldsK[0], HD, 64, HD);
    }
#else
    {
        const int r = t >> 1, c = (t & 1) * 32;
        const unsigned short* src = Qh + r * HD + c;
        unsigned short* dst = &ldsQ[r * 72 + c];
        #pragma unroll
        for (int i = 0; i < 32; i += 8)
            *(u32x4*)(dst + i) = *(const u32x4*)(src + i);
    }
#endif
    u32x4 va, vb;       // V tile registers
#if !HAVE_TDM
    u32x4 ka, kb2;      // K tile registers (fallback path)
    {
        const unsigned short* s = Kh + (size_t)vr * HD + vc;
        ka = *(const u32x4*)s; kb2 = *(const u32x4*)(s + 8);
    }
#endif
    {
        const unsigned short* s = Vh + (size_t)vr * HD + vc;
        va = *(const u32x4*)s; vb = *(const u32x4*)(s + 8);
    }
#if HAVE_TDM
    if (wave == 0) tdm_wait_tensor0();
#endif
    __syncthreads();

    v16bf aQ0 = frag_ld(&ldsQ[(wave * 16 + lm) * 72], half);
    v16bf aQ1 = frag_ld(&ldsQ[(wave * 16 + lm) * 72 + 32], half);

    v8f accO[4];
    accO[0] = {}; accO[1] = {}; accO[2] = {}; accO[3] = {};
    float m[8], l[8];
    #pragma unroll
    for (int v = 0; v < 8; ++v) { m[v] = -1e30f; l[v] = 0.0f; }

    const float SC = 0.125f * 1.44269504088896f; // 1/sqrt(64) * log2(e)

    for (int it = 0; it < KTILES; ++it) {
        const int cur = it & 1;
        unsigned short* Kb = ldsK + cur * (64 * 72);
        unsigned short* Vb = ldsV + cur * (64 * 72);

        // ---- commit staged registers for tile `it` to LDS ----
#if !HAVE_TDM
        *(u32x4*)&Kb[vr * 72 + vc]     = ka;
        *(u32x4*)&Kb[vr * 72 + vc + 8] = kb2;
#endif
        #pragma unroll
        for (int i = 0; i < 4; ++i) {   // transpose V into [d][key]
            Vb[(vc + 2 * i    ) * 72 + vr] = (unsigned short)(va[i] & 0xFFFFu);
            Vb[(vc + 2 * i + 1) * 72 + vr] = (unsigned short)(va[i] >> 16);
            Vb[(vc + 8 + 2 * i) * 72 + vr] = (unsigned short)(vb[i] & 0xFFFFu);
            Vb[(vc + 9 + 2 * i) * 72 + vr] = (unsigned short)(vb[i] >> 16);
        }
#if HAVE_TDM
        if (wave == 0) tdm_wait_tensor0();   // K tile `it` DMA complete
#endif
        __syncthreads();

        // ---- prefetch tile it+1 (DMA + global loads overlap compute) ----
        if (it + 1 < KTILES) {
            const int kbn = (it + 1) * 64;
#if HAVE_TDM
            if (wave == 0)
                tdm_load_2d_bf16(Kh + (size_t)kbn * HD,
                    (unsigned)(size_t)(const void*)&ldsK[(1 - cur) * (64 * 72)],
                    HD, 64, HD);
#else
            {
                const unsigned short* s = Kh + (size_t)(kbn + vr) * HD + vc;
                ka = *(const u32x4*)s; kb2 = *(const u32x4*)(s + 8);
            }
#endif
            const unsigned short* s = Vh + (size_t)(kbn + vr) * HD + vc;
            va = *(const u32x4*)s; vb = *(const u32x4*)(s + 8);
        }

        // ---- scores: 16(q) x 64(key), depth 64 ----
        v8f sc[4];
        #pragma unroll
        for (int ni = 0; ni < 4; ++ni) {
            v8f z = {};
            z = wmma_bf16(aQ0, frag_ld(&Kb[(ni * 16 + lm) * 72], half), z);
            z = wmma_bf16(aQ1, frag_ld(&Kb[(ni * 16 + lm) * 72 + 32], half), z);
            sc[ni] = z;
        }
        #pragma unroll
        for (int ni = 0; ni < 4; ++ni)
            #pragma unroll
            for (int v = 0; v < 8; ++v) sc[ni][v] *= SC;

        // ---- online softmax (row reductions over 16-lane N-groups) ----
        float mnew[8], scale[8];
        #pragma unroll
        for (int v = 0; v < 8; ++v) {
            float mx = fmaxf(fmaxf(sc[0][v], sc[1][v]), fmaxf(sc[2][v], sc[3][v]));
            mx = fmaxf(mx, __shfl_xor(mx, 1, 32));
            mx = fmaxf(mx, __shfl_xor(mx, 2, 32));
            mx = fmaxf(mx, __shfl_xor(mx, 4, 32));
            mx = fmaxf(mx, __shfl_xor(mx, 8, 32));
            const float mn = fmaxf(m[v], mx);
            mnew[v]  = mn;
            scale[v] = exp2f(m[v] - mn);
            m[v]     = mn;
        }
        #pragma unroll
        for (int v = 0; v < 8; ++v) {
            float sum = 0.0f;
            #pragma unroll
            for (int ni = 0; ni < 4; ++ni) {
                const float p = exp2f(sc[ni][v] - mnew[v]);
                sc[ni][v] = p;
                sum += p;
            }
            sum += __shfl_xor(sum, 1, 32);
            sum += __shfl_xor(sum, 2, 32);
            sum += __shfl_xor(sum, 4, 32);
            sum += __shfl_xor(sum, 8, 32);
            l[v] = l[v] * scale[v] + sum;
        }
        #pragma unroll
        for (int ni = 0; ni < 4; ++ni)
            #pragma unroll
            for (int v = 0; v < 8; ++v) accO[ni][v] *= scale[v];

        // ---- re-layout P (C-frag -> A-frag) through per-wave LDS ----
        unsigned short* pb = &ldsP[wave * 16 * 72];
        #pragma unroll
        for (int ni = 0; ni < 4; ++ni)
            #pragma unroll
            for (int v = 0; v < 8; ++v)
                pb[(v + half * 8) * 72 + ni * 16 + lm] = f2bf(sc[ni][v]);
        v16bf aP0 = frag_ld(&ldsP[(wave * 16 + lm) * 72], half);
        v16bf aP1 = frag_ld(&ldsP[(wave * 16 + lm) * 72 + 32], half);

        // ---- O += P @ V ----
        #pragma unroll
        for (int ni = 0; ni < 4; ++ni) {
            accO[ni] = wmma_bf16(aP0, frag_ld(&Vb[(ni * 16 + lm) * 72], half), accO[ni]);
            accO[ni] = wmma_bf16(aP1, frag_ld(&Vb[(ni * 16 + lm) * 72 + 32], half), accO[ni]);
        }
    }

    // ---- finalize: O / l, store bf16 to [B][S][H][64] ----
    const int b = bh >> 4, h = bh & 15;
    #pragma unroll
    for (int v = 0; v < 8; ++v) {
        const float inv = 1.0f / l[v];
        const int srow = qb * 128 + wave * 16 + v + half * 8;
        const size_t base = (((size_t)b * SEQ + srow) * NH + h) * HD;
        #pragma unroll
        for (int ni = 0; ni < 4; ++ni)
            O[base + ni * 16 + lm] = f2bf(accO[ni][v] * inv);
    }
}

// ---------------------------------------------------------------------------
extern "C" void kernel_launch(void* const* d_in, const int* in_sizes, int n_in,
                              void* d_out, int out_size, void* d_ws, size_t ws_size,
                              hipStream_t stream) {
    (void)in_sizes; (void)n_in; (void)out_size; (void)ws_size;
    const float* x  = (const float*)d_in[0];
    const float* Wq = (const float*)d_in[1];
    const float* bq = (const float*)d_in[2];
    const float* Wk = (const float*)d_in[3];
    const float* bk = (const float*)d_in[4];
    const float* Wv = (const float*)d_in[5];
    const float* bv = (const float*)d_in[6];
    const float* Wo = (const float*)d_in[7];
    const float* bo = (const float*)d_in[8];

    // workspace: bf16 Q,K,V [B*H][S][64]; bf16 attn-out [B][S][H*64]; bf16 x
    unsigned short* q_ws = (unsigned short*)d_ws;
    unsigned short* k_ws = q_ws + (size_t)MTOT * DM;
    unsigned short* v_ws = k_ws + (size_t)MTOT * DM;
    unsigned short* a_ws = v_ws + (size_t)MTOT * DM;
    unsigned short* x_bf = a_ws + (size_t)MTOT * DM;

    f32_to_bf16_kernel<<<dim3((MTOT * DM) / (256 * 8)), dim3(256), 0, stream>>>(x, x_bf);

    dim3 gg(DM / 64, MTOT / 128);   // (16, 32)
    gemm_bias_kernel<1><<<gg, dim3(256), 0, stream>>>(x_bf, Wq, bq, q_ws);
    gemm_bias_kernel<1><<<gg, dim3(256), 0, stream>>>(x_bf, Wk, bk, k_ws);
    gemm_bias_kernel<1><<<gg, dim3(256), 0, stream>>>(x_bf, Wv, bv, v_ws);

    dim3 ga(SEQ / 128, BSZ * NH);   // (16, 32)
    attn_kernel<<<ga, dim3(256), 0, stream>>>(q_ws, k_ws, v_ws, a_ws);

    gemm_bias_kernel<0><<<gg, dim3(256), 0, stream>>>(a_ws, Wo, bo, d_out);
}
